// LPKT_55387898250090
// MI455X (gfx1250) — matrix operand order repaired and checked by reading, not compiled, not run on hardware
//
#include <hip/hip_runtime.h>
#include <hip/hip_bf16.h>

typedef __attribute__((ext_vector_type(16))) _Float16 v16h;
typedef __attribute__((ext_vector_type(8)))  _Float16 v8h;
typedef __attribute__((ext_vector_type(8)))  float    v8f;

namespace {
constexpr int kB  = 16;    // batch
constexpr int kS  = 128;   // sequence
constexpr int kNC = 100;   // concepts
constexpr int kDK = 128;   // hidden dim
constexpr int kMR = 112;   // NC padded to 7 WMMA M-tiles
constexpr int kMT = 7;     // M tiles in main-loop GEMM
constexpr int kWAVES = 8;  // 256 threads / wave32
}

__device__ __forceinline__ float sigf(float x) { return 1.0f / (1.0f + __expf(-x)); }

__device__ __forceinline__ v16h cat16(v8h lo, v8h hi) {
  return __builtin_shufflevector(lo, hi, 0,1,2,3,4,5,6,7,8,9,10,11,12,13,14,15);
}

__global__ __launch_bounds__(256, 1)
void lpkt_kernel(const int* __restrict__ qseq, const int* __restrict__ atseq,
                 const int* __restrict__ itseq, const float* __restrict__ corr,
                 const float* __restrict__ qmat, const float* __restrict__ h0,
                 const float* __restrict__ e_w, const float* __restrict__ at_w,
                 const float* __restrict__ it_w,
                 const float* __restrict__ W1, const float* __restrict__ b1,
                 const float* __restrict__ W2, const float* __restrict__ b2,
                 const float* __restrict__ W3, const float* __restrict__ b3,
                 const float* __restrict__ W4, const float* __restrict__ b4,
                 const float* __restrict__ W5, const float* __restrict__ b5,
                 float* __restrict__ out) {
  __shared__ float    learn[kS * kDK];     // all_learning rows (64 KB)
  __shared__ float    hF[kMR * kDK];       // f32 master of h (56 KB)
  __shared__ _Float16 hH[kMR * kDK];       // f16 WMMA copy of h (28 KB)
  __shared__ _Float16 aT[16 * 384];        // phase-A staging tile (12 KB)
  __shared__ float    qe[kMR], qn[kMR];
  __shared__ float    itv[kDK], env[kDK], lgv[kDK], sv[kDK];
  __shared__ float    htl[kDK], htlN[kDK], tmpA[kDK], tmpB[kDK], red[kDK], zrow[kDK];

  const int tid  = threadIdx.x;
  const int lane = tid & 31;
  const int wave = tid >> 5;
  const int b    = blockIdx.x;
  const int row0 = lane & 15;
  const int grp  = lane >> 4;

  // ---------------- Phase A: all_learning = [e|at|corr] @ W1 + b1 (WMMA) -------------
  for (int mt = 0; mt < 8; ++mt) {
    for (int r = wave; r < 16; r += kWAVES) {
      const int t  = mt * 16 + r;
      const int q  = qseq[b * kS + t];
      const int at = atseq[b * kS + t];
      const float cv = corr[b * kS + t];
      for (int k = lane; k < kDK; k += 32) {
        aT[r * 384 + k]           = (_Float16)e_w[q * kDK + k];
        aT[r * 384 + kDK + k]     = (_Float16)at_w[at * kDK + k];
        aT[r * 384 + 2 * kDK + k] = (_Float16)cv;
      }
    }
    __syncthreads();
    {
      const int colA = wave * 16 + row0;
      v8f c = {};
      // process K in groups of 4: batch the LDS loads, then the WMMA chain,
      // so ds_load latency overlaps the matrix pipe instead of stalling it.
      #pragma unroll
      for (int kg = 0; kg < 3; ++kg) {
        v16h a4[4];
        #pragma unroll
        for (int j4 = 0; j4 < 4; ++j4) {
          const int kt = kg * 4 + j4;
          const _Float16* ap = &aT[row0 * 384 + kt * 32 + grp * 8];
          a4[j4] = cat16(*(const v8h*)ap, *(const v8h*)(ap + 16));
        }
        #pragma unroll
        for (int j4 = 0; j4 < 4; ++j4) {
          const int kt = kg * 4 + j4;
          const int kb = kt * 32 + grp * 8;
          v16h bf;
          #pragma unroll
          for (int j = 0; j < 16; ++j) bf[j] = (_Float16)W1[(kb + j + (j >> 3) * 8) * kDK + colA];
          c = __builtin_amdgcn_wmma_f32_16x16x32_f16(false, a4[j4], false, bf, (short)0, c, false, false);
        }
      }
      #pragma unroll
      for (int r = 0; r < 8; ++r) {
        const int tr = mt * 16 + grp * 8 + r;
        learn[tr * kDK + colA] = c[r] + b1[colA];
      }
    }
    __syncthreads();
  }

  // ---------------- Phase B: init h (f32 + f16), zrow, h_tilde0 ----------------------
  for (int idx = tid; idx < kMR * kDK; idx += 256) {
    const int m = idx >> 7;
    const float v = (m < kNC) ? h0[idx] : 0.0f;
    hF[idx] = v;
    hH[idx] = (_Float16)v;
  }
  if (tid < kDK) zrow[tid] = 0.0f;
  if (tid == 0) out[b * kS] = 0.0f;   // pred[:,0] = 0
  __syncthreads();
  if (tid < kDK) {
    const int q0 = qseq[b * kS];
    float a = 0.0f;
    for (int m = 0; m < kNC; ++m) a = fmaf(qmat[q0 * kNC + m], hF[m * kDK + tid], a);
    htl[tid] = a;                      // h_tilde0
  }

  // Pre-stage W4[0:128,:] B-fragments in registers (reused across 127 steps)
  v16h bW4[4];
  {
    const int col = wave * 16 + row0;
    #pragma unroll
    for (int kt = 0; kt < 4; ++kt) {
      const int kb = kt * 32 + grp * 8;
      #pragma unroll
      for (int j = 0; j < 16; ++j) bW4[kt][j] = (_Float16)W4[(kb + j + (j >> 3) * 8) * kDK + col];
    }
  }
  __syncthreads();

  // ---------------- Main recurrence: t = 0 .. S-2 ------------------------------------
  for (int t = 0; t < kS - 1; ++t) {
    const int qt  = qseq[b * kS + t];
    const int qt1 = qseq[b * kS + t + 1];
    const int itt = itseq[b * kS + t];
    if (tid < kDK) {
      itv[tid]  = it_w[itt * kDK + tid];
      env[tid]  = e_w[qt1 * kDK + tid];
      htlN[tid] = 0.0f;
    }
    if (tid < kMR) {
      qe[tid] = (tid < kNC) ? qmat[qt * kNC + tid] : 0.0f;
      qn[tid] = (tid < kNC) ? qmat[qt1 * kNC + tid] : 0.0f;
    }
    if (tid == 0 && t + 2 < kS) __builtin_prefetch(&e_w[qseq[b * kS + t + 2] * kDK], 0, 0);
    __syncthreads();

    // gate pre-activations: x4 = [learn_{t-1} | it | learn_t | h_tilde] @ {W2,W3}
    {
      const int n = tid & (kDK - 1);
      const float* Wm = (tid < kDK) ? W2 : W3;
      float a = (tid < kDK) ? b2[n] : b3[n];
      const float* lp = (t > 0) ? &learn[(t - 1) * kDK] : zrow;
      const float* lc = &learn[t * kDK];
      #pragma unroll 4
      for (int k = 0; k < kDK; ++k) a = fmaf(lp[k],  Wm[k * kDK + n], a);
      #pragma unroll 4
      for (int k = 0; k < kDK; ++k) a = fmaf(itv[k], Wm[(kDK + k) * kDK + n], a);
      #pragma unroll 4
      for (int k = 0; k < kDK; ++k) a = fmaf(lc[k],  Wm[(2 * kDK + k) * kDK + n], a);
      #pragma unroll 4
      for (int k = 0; k < kDK; ++k) a = fmaf(htl[k], Wm[(3 * kDK + k) * kDK + n], a);
      if (tid < kDK) tmpA[n] = a; else tmpB[n] = a;
    }
    __syncthreads();
    if (tid < kDK) {
      const float lg = tanhf(tmpA[tid]);
      const float gl = sigf(tmpB[tid]);
      lgv[tid] = gl * (lg + 1.0f) * 0.5f;   // LG
    }
    __syncthreads();

    // row-constant part of gamma_f input: LG@W4[128:256] + it@W4[256:384] + b4
    {
      const int n = tid & (kDK - 1);
      const float* vec = (tid < kDK) ? lgv : itv;
      const int ro = (tid < kDK) ? kDK : 2 * kDK;
      float a = 0.0f;
      #pragma unroll 4
      for (int k = 0; k < kDK; ++k) a = fmaf(vec[k], W4[(ro + k) * kDK + n], a);
      if (tid < kDK) tmpA[n] = a; else tmpB[n] = a;
    }
    __syncthreads();
    if (tid < kDK) sv[tid] = tmpA[tid] + tmpB[tid] + b4[tid];
    __syncthreads();

    // WMMA: per-row part  hH(112x128) @ W4[0:128,:]
    // Software-pipelined: preload next M-tile's A-fragments during current WMMA chain.
    v8f acc[kMT];
    {
      v16h aCur[4], aNxt[4];
      #pragma unroll
      for (int kt = 0; kt < 4; ++kt) {
        const _Float16* ap = &hH[row0 * kDK + kt * 32 + grp * 8];
        aCur[kt] = cat16(*(const v8h*)ap, *(const v8h*)(ap + 16));
      }
      #pragma unroll
      for (int mt = 0; mt < kMT; ++mt) {
        if (mt + 1 < kMT) {
          #pragma unroll
          for (int kt = 0; kt < 4; ++kt) {
            const _Float16* ap = &hH[((mt + 1) * 16 + row0) * kDK + kt * 32 + grp * 8];
            aNxt[kt] = cat16(*(const v8h*)ap, *(const v8h*)(ap + 16));
          }
        }
        v8f c = {};
        #pragma unroll
        for (int kt = 0; kt < 4; ++kt)
          c = __builtin_amdgcn_wmma_f32_16x16x32_f16(false, aCur[kt], false, bW4[kt], (short)0, c, false, false);
        acc[mt] = c;
        #pragma unroll
        for (int kt = 0; kt < 4; ++kt) aCur[kt] = aNxt[kt];
      }
    }
    __syncthreads();   // all hH reads complete before in-place update

    // h_new = q_e*LG + sigmoid(pre + s)*h ; accumulate h_tilde = q_next . h_new
    {
      const int col = wave * 16 + row0;
      const float LGn = lgv[col];
      const float sn  = sv[col];
      float hpart = 0.0f;
      #pragma unroll
      for (int mt = 0; mt < kMT; ++mt) {
        #pragma unroll
        for (int r = 0; r < 8; ++r) {
          const int m = mt * 16 + grp * 8 + r;
          const float gf = sigf(acc[mt][r] + sn);
          const float hn = qe[m] * LGn + gf * hF[m * kDK + col];
          hF[m * kDK + col] = hn;
          hH[m * kDK + col] = (_Float16)hn;
          hpart = fmaf(qn[m], hn, hpart);
        }
      }
      atomicAdd(&htlN[col], hpart);   // ds_add_f32; 2 lanes contribute per column
    }
    __syncthreads();

    // y = mean(sigmoid([e_next | h_tilde] @ W5 + b5))
    if (tid < kDK) {
      const int n = tid;
      float a = b5[n];
      #pragma unroll 4
      for (int k = 0; k < kDK; ++k) a = fmaf(env[k],  W5[k * kDK + n], a);
      #pragma unroll 4
      for (int k = 0; k < kDK; ++k) a = fmaf(htlN[k], W5[(kDK + k) * kDK + n], a);
      red[n] = sigf(a);
      htl[n] = htlN[n];               // carry h_tilde for next step
    }
    __syncthreads();
    #pragma unroll
    for (int off = 64; off >= 1; off >>= 1) {
      if (tid < off) red[tid] += red[tid + off];
      __syncthreads();
    }
    if (tid == 0) out[b * kS + t + 1] = red[0] * (1.0f / 128.0f);
  }
}

extern "C" void kernel_launch(void* const* d_in, const int* in_sizes, int n_in,
                              void* d_out, int out_size, void* d_ws, size_t ws_size,
                              hipStream_t stream) {
  (void)in_sizes; (void)n_in; (void)out_size; (void)d_ws; (void)ws_size;
  lpkt_kernel<<<kB, 256, 0, stream>>>(
      (const int*)d_in[0],    // question_seq
      (const int*)d_in[1],    // use_time_seq
      (const int*)d_in[2],    // interval_time_seq
      (const float*)d_in[3],  // correct_seq
      (const float*)d_in[4],  // q_matrix
      (const float*)d_in[5],  // h0
      (const float*)d_in[6],  // e_w
      (const float*)d_in[7],  // at_w
      (const float*)d_in[8],  // it_w
      (const float*)d_in[9],  (const float*)d_in[10],   // W1,b1
      (const float*)d_in[11], (const float*)d_in[12],   // W2,b2
      (const float*)d_in[13], (const float*)d_in[14],   // W3,b3
      (const float*)d_in[15], (const float*)d_in[16],   // W4,b4
      (const float*)d_in[17], (const float*)d_in[18],   // W5,b5
      (float*)d_out);
}
